// DIFFormerConv_18150531793318
// MI455X (gfx1250) — matrix operand order repaired and checked by reading, not compiled, not run on hardware
//
#include <hip/hip_runtime.h>
#include <hip/hip_bf16.h>

// DIFFormerConv for MI455X / gfx1250 (wave32, WMMA f32_16x16x32_f16,
// async global->LDS staging with ASYNCcnt, double-buffered GCN GEMM).
// B=8, C=256, N=4096, H=4, D=64, HD=256.

#define Bb 8
#define Cc 256
#define Nn 4096
#define Hh 4
#define Dd 64
#define HD 256

typedef __attribute__((ext_vector_type(4)))  _Float16 v4h;
typedef __attribute__((ext_vector_type(8)))  _Float16 v8h;
typedef __attribute__((ext_vector_type(16))) _Float16 v16h;
typedef __attribute__((ext_vector_type(8)))  float    v8f;

__device__ __forceinline__ v8f vzero8() {
    v8f z;
#pragma unroll
    for (int i = 0; i < 8; ++i) z[i] = 0.0f;
    return z;
}

__device__ __forceinline__ v8f wmma_f16(v16h a, v16h b, v8f c) {
    // D = A(16x32 f16) * B(32x16 f16) + C(16x16 f32)
    return __builtin_amdgcn_wmma_f32_16x16x32_f16(false, a, false, b,
                                                  (short)0, c, false, false);
}

// Async 16B global -> LDS copy (GLOBAL_LOAD_ASYNC_TO_LDS_B128, ASYNCcnt).
// LDS operand is the 32-bit LDS byte offset (= low 32 bits of the generic
// pointer per the flat->LDS aperture mapping, ISA 10.2).
__device__ __forceinline__ void async_copy16(const _Float16* lds_dst,
                                             const _Float16* gsrc) {
    const unsigned loff = (unsigned)(size_t)lds_dst;
    asm volatile("global_load_async_to_lds_b128 %0, %1, off"
                 :: "v"(loff), "v"((unsigned long long)(size_t)gsrc)
                 : "memory");
}
__device__ __forceinline__ void wait_async0() {
    asm volatile("s_wait_asynccnt 0x0" ::: "memory");
}

// A fragment (16x32 f16, row-major LDS tile, row stride `ld` halves).
// ISA layout: lane m=l&15, half=l>>4; VGPR0-3 hold K = half*8 + 0..7,
// VGPR4-7 hold K = 16 + half*8 + 0..7  -> two b128 LDS loads per lane.
__device__ __forceinline__ v16h frag_a(const _Float16* base, int ld) {
    const int lane = threadIdx.x & 31;
    const int m = lane & 15, h8 = (lane >> 4) * 8;
    const _Float16* p = base + m * ld + h8;
    v8h lo = *(const v8h*)(p);
    v8h hi = *(const v8h*)(p + 16);
    return __builtin_shufflevector(lo, hi, 0,1,2,3,4,5,6,7,8,9,10,11,12,13,14,15);
}

// B fragment (32x16 f16) from a COLUMN-MAJOR LDS tile b[col][k], col stride `ld`.
// ISA layout: lane col=l&15, half=l>>4; VGPR0-3 hold K = half*8 + 0..7,
// VGPR4-7 hold K = 16 + half*8 + 0..7.
__device__ __forceinline__ v16h frag_b(const _Float16* baseColMajor, int ld) {
    const int lane = threadIdx.x & 31;
    const int n = lane & 15, h16 = (lane >> 4) * 8;
    const _Float16* p = baseColMajor + n * ld + h16;
    v8h lo = *(const v8h*)(p);
    v8h hi = *(const v8h*)(p + 16);
    return __builtin_shufflevector(lo, hi, 0,1,2,3,4,5,6,7,8,9,10,11,12,13,14,15);
}

// ---------------- adj_n = (adj + I) / rowsum, f32 -> f16, one fused pass.
// One block per row: row fits in registers (16 f32/thread), reduce, rescale.
__global__ __launch_bounds__(256)
void adjn_kernel(const float* __restrict__ adj, _Float16* __restrict__ adjn) {
    __shared__ float red[256];
    const int row = blockIdx.x, t = threadIdx.x;
    const float4* p = (const float4*)(adj + (size_t)row * Nn);
    float4 r4[4];
    float s = 0.f;
#pragma unroll
    for (int i = 0; i < 4; ++i) {
        r4[i] = p[t + i * 256];
        s += (r4[i].x + r4[i].y) + (r4[i].z + r4[i].w);
    }
    red[t] = s; __syncthreads();
    for (int off = 128; off > 0; off >>= 1) {
        if (t < off) red[t] += red[t + off];
        __syncthreads();
    }
    const float inv = 1.0f / (red[0] + 1.0f);   // +1 = self loop on diagonal
#pragma unroll
    for (int i = 0; i < 4; ++i) {
        const int col = (t + i * 256) * 4;
        const float e0 = r4[i].x + (row == col + 0 ? 1.f : 0.f);
        const float e1 = r4[i].y + (row == col + 1 ? 1.f : 0.f);
        const float e2 = r4[i].z + (row == col + 2 ? 1.f : 0.f);
        const float e3 = r4[i].w + (row == col + 3 ? 1.f : 0.f);
        v4h h;
        h[0] = (_Float16)(e0 * inv); h[1] = (_Float16)(e1 * inv);
        h[2] = (_Float16)(e2 * inv); h[3] = (_Float16)(e3 * inv);
        *(v4h*)(adjn + (size_t)row * Nn + col) = h;
    }
}

// --------------------------------------------- fused QKV projection (WMMA)
// type 0: q -> normalize -> qn_h
// type 1: k -> normalize -> kn_h, + kssum atomic reduce
// type 2: v -> v_h, + vsum atomic reduce, + vbarT (head mean, [b*64+d][n])
__global__ __launch_bounds__(256)
void proj_kernel(const float* __restrict__ xq, const float* __restrict__ xs,
                 const float* __restrict__ Wq, const float* __restrict__ bq,
                 const float* __restrict__ Wk, const float* __restrict__ bk,
                 const float* __restrict__ Wv, const float* __restrict__ bv,
                 _Float16* __restrict__ qn_h, _Float16* __restrict__ kn_h,
                 _Float16* __restrict__ v_h,  _Float16* __restrict__ vbarT_h,
                 float* __restrict__ kssum, float* __restrict__ vsum) {
    const int n0 = blockIdx.x * 32;
    const int b  = blockIdx.y;
    const int type = blockIdx.z;
    const float* x    = (type == 0) ? xq : xs;
    const float* W    = (type == 0) ? Wq : (type == 1) ? Wk : Wv;
    const float* bias = (type == 0) ? bq : (type == 1) ? bk : bv;

    __shared__ __align__(16) _Float16 x_lds[32 * 32];       // [node][k]
    __shared__ __align__(16) _Float16 w_lds[HD * 32];       // [col][k] col-major
    __shared__ __align__(16) float    acc_lds[32 * HD];     // [node][col]

    const int t = threadIdx.x, lane = t & 31, wv = t >> 5;  // 8 waves
    const int node_sub = wv & 1, col_q = wv >> 1;           // 2x16 nodes, 4x64 cols

    v8f acc[4];
#pragma unroll
    for (int i = 0; i < 4; ++i) acc[i] = vzero8();

    for (int c0 = 0; c0 < Cc; c0 += 32) {
        // stage x tile (32 nodes x 32 k), f32 -> f16 (x is [B,C,N])
#pragma unroll
        for (int i = 0; i < 4; ++i) {
            int e = t + i * 256;
            int n = e & 31, c = e >> 5;
            x_lds[n * 32 + c] =
                (_Float16)x[(size_t)(b * Cc + c0 + c) * Nn + n0 + n];
        }
        // stage W tile (256 cols x 32 k), col-major in LDS; thread t = one col
        {
            const float4* wp = (const float4*)(W + (size_t)t * Cc + c0);
#pragma unroll
            for (int i = 0; i < 8; ++i) {
                float4 v4 = wp[i];
                _Float16* d = w_lds + t * 32 + i * 4;
                d[0] = (_Float16)v4.x; d[1] = (_Float16)v4.y;
                d[2] = (_Float16)v4.z; d[3] = (_Float16)v4.w;
            }
        }
        __syncthreads();
        v16h af = frag_a(x_lds + node_sub * 16 * 32, 32);
#pragma unroll
        for (int tt = 0; tt < 4; ++tt) {
            v16h bf = frag_b(w_lds + (col_q * 64 + tt * 16) * 32, 32);
            acc[tt] = wmma_f16(af, bf, acc[tt]);
        }
        __syncthreads();
    }

    // spill accumulators to LDS (C/D layout: lane l -> m=r+8*(l>>4), col=l&15)
    {
        const int m8 = (lane >> 4) * 8, cl = lane & 15;
#pragma unroll
        for (int tt = 0; tt < 4; ++tt)
#pragma unroll
            for (int r = 0; r < 8; ++r)
                acc_lds[(node_sub * 16 + m8 + r) * HD + col_q * 64 + tt * 16 + cl]
                    = acc[tt][r];
    }
    __syncthreads();

    if (type <= 1) {
        if (t < 128) {  // one thread per (node, head)
            const int node = t >> 2, head = t & 3;
            float ss = 0.f;
            for (int d = 0; d < Dd; ++d) {
                float v = acc_lds[node * HD + head * Dd + d] + bias[head * Dd + d];
                ss += v * v;
            }
            const float rs = rsqrtf(ss);
            _Float16* outp = ((type == 0) ? qn_h : kn_h)
                + ((size_t)(b * Nn + n0 + node) * HD + head * Dd);
            for (int d = 0; d < Dd; ++d) {
                float v = acc_lds[node * HD + head * Dd + d] + bias[head * Dd + d];
                float nv = v * rs;
                outp[d] = (_Float16)nv;
                acc_lds[node * HD + head * Dd + d] = nv;
            }
        }
        __syncthreads();
        if (type == 1) {        // kssum: one atomic per (b, h, d) per WG
            float s = 0.f;
            for (int node = 0; node < 32; ++node) s += acc_lds[node * HD + t];
            atomicAdd(kssum + b * HD + t, s);
        }
    } else {
        if (t < 128) {
            const int node = t >> 2, head = t & 3;
            _Float16* outp = v_h + ((size_t)(b * Nn + n0 + node) * HD + head * Dd);
            for (int d = 0; d < Dd; ++d) {
                float v = acc_lds[node * HD + head * Dd + d] + bias[head * Dd + d];
                outp[d] = (_Float16)v;
                acc_lds[node * HD + head * Dd + d] = v;
            }
        }
        __syncthreads();
        {   // vsum
            float s = 0.f;
            for (int node = 0; node < 32; ++node) s += acc_lds[node * HD + t];
            atomicAdd(vsum + b * HD + t, s);
        }
        {   // vbarT = mean over heads, stored [b*64+d][m] so the GCN B tile
            // is a contiguous 64B run per column -> async b128 staging.
            const int node = t & 31, d0 = (t >> 5) * 8;
#pragma unroll
            for (int j = 0; j < 8; ++j) {
                const int d = d0 + j;
                float m = 0.25f * (acc_lds[node * HD + d]
                                 + acc_lds[node * HD + 64 + d]
                                 + acc_lds[node * HD + 128 + d]
                                 + acc_lds[node * HD + 192 + d]);
                vbarT_h[(size_t)(b * Dd + d) * Nn + n0 + node] = (_Float16)m;
            }
        }
    }
}

// ------------------------------------------------ kvs[b,h] = kn^T v (64x64)
__global__ __launch_bounds__(512)
void kvs_kernel(const _Float16* __restrict__ kn_h, const _Float16* __restrict__ v_h,
                float* __restrict__ kvs) {
    const int hh = blockIdx.x, b = blockIdx.y;
    __shared__ __align__(16) _Float16 a_ldsT[Dd * 32];  // kn^T: [d1][n]
    __shared__ __align__(16) _Float16 b_ldsT[Dd * 32];  // v^T:  [d2][n] (col-major B)
    const int t = threadIdx.x, lane = t & 31, wv = t >> 5;   // 16 waves
    const int mt = wv >> 2, ct = wv & 3;
    v8f acc = vzero8();

    for (int n0 = 0; n0 < Nn; n0 += 32) {
        const int n = t >> 4, d0 = (t & 15) * 4;
        const _Float16* kp = kn_h + ((size_t)(b * Nn + n0 + n) * HD + hh * Dd + d0);
        const _Float16* vp = v_h  + ((size_t)(b * Nn + n0 + n) * HD + hh * Dd + d0);
#pragma unroll
        for (int i = 0; i < 4; ++i) {
            a_ldsT[(d0 + i) * 32 + n] = kp[i];
            b_ldsT[(d0 + i) * 32 + n] = vp[i];
        }
        __syncthreads();
        v16h af = frag_a(a_ldsT + mt * 16 * 32, 32);
        v16h bf = frag_b(b_ldsT + ct * 16 * 32, 32);
        acc = wmma_f16(af, bf, acc);
        __syncthreads();
    }
    const int m8 = (lane >> 4) * 8, cl = lane & 15;
#pragma unroll
    for (int r = 0; r < 8; ++r)
        kvs[(size_t)((b * Hh + hh) * Dd + mt * 16 + m8 + r) * Dd + ct * 16 + cl]
            = acc[r];
}

// --------------------------- attention epilogue: (qn@kvs + vsum)/denom, head mean
__global__ __launch_bounds__(256)
void attn_kernel(const _Float16* __restrict__ qn_h, const float* __restrict__ kvs,
                 const float* __restrict__ kssum, const float* __restrict__ vsum,
                 float* __restrict__ out) {
    const int n0 = blockIdx.x * 32, b = blockIdx.y;
    __shared__ __align__(16) _Float16 qn_lds[32 * HD];       // [node][c]
    __shared__ __align__(16) _Float16 kvsT[Hh * Dd * Dd];    // [h][d2][d1] col-major B
    __shared__ float denom_lds[32 * Hh];
    __shared__ float outacc[32 * Dd];
    const int t = threadIdx.x, lane = t & 31, wv = t >> 5;

    {   // stage qn: contiguous 16KB, async global->LDS DMA (no VGPR roundtrip)
        const _Float16* src = qn_h + (size_t)(b * Nn + n0) * HD;
#pragma unroll
        for (int i = 0; i < 4; ++i) {
            const int c = t + i * 256;
            async_copy16(qn_lds + c * 8, src + c * 8);
        }
    }
    for (int idx = t; idx < Hh * Dd * Dd; idx += 256) {   // stage kvs^T f32->f16
        const int sh = idx >> 12, rem = idx & 4095, d1 = rem >> 6, d2 = rem & 63;
        kvsT[sh * 4096 + d2 * Dd + d1] =
            (_Float16)kvs[(size_t)((b * Hh + sh) * Dd + d1) * Dd + d2];
    }
    for (int i = t; i < 32 * Dd; i += 256) outacc[i] = 0.f;
    wait_async0();
    __syncthreads();

    if (t < 128) {   // denom per (node, head)
        const int node = t >> 2, dh = t & 3;
        float s = 0.f;
        for (int d = 0; d < Dd; ++d)
            s += (float)qn_lds[node * HD + dh * Dd + d] * kssum[b * HD + dh * Dd + d];
        denom_lds[node * Hh + dh] = s + (float)Nn;
    }
    __syncthreads();

    const int nt = wv >> 2, hh = wv & 3;   // 2 node tiles x 4 heads
    v8f acc[4];
#pragma unroll
    for (int i = 0; i < 4; ++i) acc[i] = vzero8();
#pragma unroll
    for (int ks = 0; ks < 2; ++ks) {       // K = 64 = 2 x 32
        v16h af = frag_a(qn_lds + (nt * 16) * HD + hh * Dd + ks * 32, HD);
#pragma unroll
        for (int ct = 0; ct < 4; ++ct) {
            v16h bf = frag_b(kvsT + hh * 4096 + (ct * 16) * Dd + ks * 32, Dd);
            acc[ct] = wmma_f16(af, bf, acc[ct]);
        }
    }
    const int m8 = (lane >> 4) * 8, cl = lane & 15;
#pragma unroll
    for (int ct = 0; ct < 4; ++ct)
#pragma unroll
        for (int r = 0; r < 8; ++r) {
            const int node = nt * 16 + m8 + r, d = ct * 16 + cl;
            float val = (acc[ct][r] + vsum[b * HD + hh * Dd + d]) * 0.25f
                        / denom_lds[node * Hh + hh];
            atomicAdd(&outacc[node * Dd + d], val);   // ds_add_f32, head mean
        }
    __syncthreads();
    for (int i = t; i < 32 * Dd; i += 256) {
        const int n = i & 31, d = i >> 5;
        out[(size_t)(b * Dd + d) * Nn + n0 + n] = outacc[n * Dd + d];
    }
}

// -------------------- GCN GEMM: out[b,d,n] += sum_m adj_n[n,m] * vbar[b,m,d]
// A = adjn (f16 row-major [n][m], L2-resident at 33.5MB), B = vbarT [col][m].
// Double-buffered tiles staged with GLOBAL_LOAD_ASYNC_TO_LDS_B128.
__device__ __forceinline__ void gcn_stage(const _Float16* __restrict__ adjn,
                                          const _Float16* __restrict__ vbarT,
                                          _Float16* a_buf, _Float16* b_buf,
                                          int nblk, int cblk, int m0, int t) {
#pragma unroll
    for (int i = 0; i < 2; ++i) {
        const int c = t + i * 256;          // 512 chunks of 16B per matrix
        const int row = c >> 2, q = c & 3;  // 128 rows/cols x 4 x 16B
        async_copy16(a_buf + row * 32 + q * 8,
                     adjn + (size_t)(nblk + row) * Nn + m0 + q * 8);
        async_copy16(b_buf + row * 32 + q * 8,
                     vbarT + (size_t)(cblk + row) * Nn + m0 + q * 8);
    }
}

__global__ __launch_bounds__(256)
void gcn_kernel(const _Float16* __restrict__ adjn,
                const _Float16* __restrict__ vbarT, float* __restrict__ out) {
    const int nblk = blockIdx.x * 128;
    const int cblk = blockIdx.y * 128;          // col = b*64 + d, 512 cols total
    __shared__ __align__(16) _Float16 a_lds[2][128 * 32];   // [n][m] row-major
    __shared__ __align__(16) _Float16 b_lds[2][128 * 32];   // [col][m] col-major
    const int t = threadIdx.x, lane = t & 31, wv = t >> 5;
    const int nsub = wv & 1, csub = wv >> 1;    // 2x64 rows, 4x32 cols per wave

    v8f acc[4][2];
#pragma unroll
    for (int i = 0; i < 4; ++i)
#pragma unroll
        for (int j = 0; j < 2; ++j) acc[i][j] = vzero8();

    gcn_stage(adjn, vbarT, a_lds[0], b_lds[0], nblk, cblk, 0, t);   // prologue

    for (int it = 0; it < Nn / 32; ++it) {
        wait_async0();          // own async copies done
        __syncthreads();        // everyone's copies visible; prev reads retired
        if (it + 1 < Nn / 32)   // overlap next tile's DMA with this tile's WMMA
            gcn_stage(adjn, vbarT, a_lds[(it + 1) & 1], b_lds[(it + 1) & 1],
                      nblk, cblk, (it + 1) * 32, t);
        const _Float16* ab = a_lds[it & 1];
        const _Float16* bb = b_lds[it & 1];
        v16h af[4], bf[2];
#pragma unroll
        for (int i = 0; i < 4; ++i)
            af[i] = frag_a(ab + (nsub * 64 + i * 16) * 32, 32);
#pragma unroll
        for (int j = 0; j < 2; ++j)
            bf[j] = frag_b(bb + (csub * 32 + j * 16) * 32, 32);
#pragma unroll
        for (int i = 0; i < 4; ++i)
#pragma unroll
            for (int j = 0; j < 2; ++j)
                acc[i][j] = wmma_f16(af[i], bf[j], acc[i][j]);
    }

    const int m8 = (lane >> 4) * 8, cl = lane & 15;
#pragma unroll
    for (int i = 0; i < 4; ++i)
#pragma unroll
        for (int j = 0; j < 2; ++j) {
            const int colg = cblk + csub * 32 + j * 16 + cl;
            const int ob = colg >> 6, od = colg & 63;
            float* p = out + (size_t)(ob * Dd + od) * Nn
                           + nblk + nsub * 64 + i * 16 + m8;
#pragma unroll
            for (int r = 0; r < 8; ++r) p[r] += acc[i][j][r];   // += attn part
        }
}

extern "C" void kernel_launch(void* const* d_in, const int* in_sizes, int n_in,
                              void* d_out, int out_size, void* d_ws, size_t ws_size,
                              hipStream_t stream) {
    (void)in_sizes; (void)n_in; (void)out_size; (void)ws_size;
    const float* xq  = (const float*)d_in[0];
    const float* xs  = (const float*)d_in[1];
    const float* adj = (const float*)d_in[2];
    const float* Wq  = (const float*)d_in[3];
    const float* bq  = (const float*)d_in[4];
    const float* Wk  = (const float*)d_in[5];
    const float* bk  = (const float*)d_in[6];
    const float* Wv  = (const float*)d_in[7];
    const float* bv  = (const float*)d_in[8];
    float* out = (float*)d_out;

    char* ws = (char*)d_ws;
    size_t off = 0;
    auto alloc = [&](size_t bytes) -> void* {
        void* p = ws + off;
        off = (off + bytes + 255) & ~(size_t)255;
        return p;
    };
    _Float16* qn_h    = (_Float16*)alloc((size_t)Bb * Nn * HD * 2);
    _Float16* kn_h    = (_Float16*)alloc((size_t)Bb * Nn * HD * 2);
    _Float16* v_h     = (_Float16*)alloc((size_t)Bb * Nn * HD * 2);
    _Float16* vbarT_h = (_Float16*)alloc((size_t)Bb * Dd * Nn * 2);
    _Float16* adjn_h  = (_Float16*)alloc((size_t)Nn * Nn * 2);
    float*    kvs     = (float*)alloc((size_t)Bb * Hh * Dd * Dd * 4);
    float*    kssum   = (float*)alloc((size_t)Bb * HD * 4);
    float*    vsum    = (float*)alloc((size_t)Bb * HD * 4);

    hipMemsetAsync(kssum, 0, (size_t)Bb * HD * 4, stream);
    hipMemsetAsync(vsum,  0, (size_t)Bb * HD * 4, stream);

    adjn_kernel<<<Nn, 256, 0, stream>>>(adj, adjn_h);
    proj_kernel<<<dim3(Nn / 32, Bb, 3), 256, 0, stream>>>(
        xq, xs, Wq, bq, Wk, bk, Wv, bv, qn_h, kn_h, v_h, vbarT_h, kssum, vsum);
    kvs_kernel<<<dim3(Hh, Bb), 512, 0, stream>>>(kn_h, v_h, kvs);
    attn_kernel<<<dim3(Nn / 32, Bb), 256, 0, stream>>>(qn_h, kvs, kssum, vsum, out);
    gcn_kernel<<<dim3(Nn / 128, (Bb * Dd) / 128), 256, 0, stream>>>(
        adjn_h, vbarT_h, out);
}
